// SAM_LSTMCell_403726926424
// MI455X (gfx1250) — matrix operand order repaired and checked by reading, not compiled, not run on hardware
//
#include <hip/hip_runtime.h>
#include <hip/hip_bf16.h>
#include <math.h>

// ---------------- problem constants ----------------
#define B_ROWS   16384
#define H_DIM    128
#define IN_DIM   128
#define GATES    640          // 5*H
#define K_TOT    256          // IN + H (concatenated GEMM K)
#define NMEM     1106         // 1100 + 3*2
#define KNB      25           // (2*2+1)^2 neighborhood
#define SPW      2

typedef __attribute__((ext_vector_type(16))) __bf16 bf16x16;
typedef __attribute__((ext_vector_type(8)))  __bf16 bf16x8;
typedef __attribute__((ext_vector_type(4)))  __bf16 bf16x4;
typedef __attribute__((ext_vector_type(8)))  float  f32x8;
typedef __attribute__((ext_vector_type(4)))  float  f32x4;

// fast activations on v_exp_f32 + v_rcp_f32; saturate correctly at +/-inf
__device__ __forceinline__ float fast_sigmoid(float v) {
    return __builtin_amdgcn_rcpf(1.0f + __expf(-v));
}
__device__ __forceinline__ float fast_tanh(float v) {
    return 1.0f - 2.0f * __builtin_amdgcn_rcpf(__expf(2.0f * v) + 1.0f);
}
// branchless unified gate activation: a=1 -> sigmoid, a=2 -> tanh (= 2*sigmoid(2x)-1)
__device__ __forceinline__ float gate_act(float g, float a) {
    return a * __builtin_amdgcn_rcpf(1.0f + __expf(-a * g)) - (a - 1.0f);
}

// Build a 16-element bf16 fragment from two contiguous 8-element chunks.
__device__ __forceinline__ bf16x16 make_frag(const __bf16* p0, const __bf16* p1) {
    bf16x8 lo = *(const bf16x8*)p0;
    bf16x8 hi = *(const bf16x8*)(p1);
    return __builtin_shufflevector(lo, hi, 0,1,2,3,4,5,6,7,8,9,10,11,12,13,14,15);
}

// ---------------- K1: weight conversion to bf16 ----------------
__global__ void convert_weights_kernel(const float* __restrict__ w_ih,
                                       const float* __restrict__ w_hh,
                                       const float* __restrict__ attn_w,
                                       __bf16* __restrict__ w2,
                                       __bf16* __restrict__ attn_wbf) {
    int i4 = blockIdx.x * 256 + threadIdx.x;      // one float4 per thread
    if (i4 < GATES * K_TOT / 4) {
        int i = i4 * 4;
        int n = i >> 8, k = i & 255;              // k multiple of 4 -> same source
        const float* src = (k < IN_DIM) ? (w_ih + (size_t)n * IN_DIM + k)
                                        : (w_hh + (size_t)n * H_DIM + (k - IN_DIM));
        f32x4 v = *(const f32x4*)src;
        *(bf16x4*)(w2 + i) = __builtin_convertvector(v, bf16x4);
    }
    if (i4 < H_DIM * K_TOT / 4) {
        f32x4 v = *(const f32x4*)(attn_w + i4 * 4);
        *(bf16x4*)(attn_wbf + i4 * 4) = __builtin_convertvector(v, bf16x4);
    }
}

// ---------------- shared GEMM tile config ----------------
#define GT_TM 128
#define GT_TN 64
#define GT_TK 32
#define LDA   (GT_TK + 8)     // pad to stagger LDS bank groups

// stage A (f32 -> bf16): 128x32 tile; 8 floats per thread-step -> one 16B DS store
__device__ __forceinline__ void stage_a(const float* __restrict__ srcA, int m0,
                                        __bf16 (*a_s)[LDA], int tid) {
#pragma unroll
    for (int t = 0; t < 2; ++t) {                 // 512 8-float chunks / 256 threads
        int i  = tid + t * 256;
        int r  = i >> 2;                          // 4 chunks per 32-wide row
        int c8 = (i & 3) * 8;
        const float* p = srcA + (size_t)(m0 + r) * 128 + c8;
        f32x4 v0 = *(const f32x4*)p;
        f32x4 v1 = *(const f32x4*)(p + 4);
        bf16x4 b0 = __builtin_convertvector(v0, bf16x4);
        bf16x4 b1 = __builtin_convertvector(v1, bf16x4);
        *(bf16x8*)(&a_s[r][c8]) = __builtin_shufflevector(b0, b1, 0,1,2,3,4,5,6,7);
    }
}
// stage B (bf16): 64x32 tile, one 16B load/store per thread
__device__ __forceinline__ void stage_b(const __bf16* __restrict__ srcB, int n0, int kk,
                                        __bf16 (*b_s)[LDA], int tid) {
    int nr = tid >> 2;
    int c8 = (tid & 3) * 8;
    bf16x8 v = *(const bf16x8*)(srcB + (size_t)(n0 + nr) * K_TOT + kk + c8);
    *(bf16x8*)(&b_s[nr][c8]) = v;
}

// ---------------- K2: gates GEMM (bf16 WMMA) + fused branchless activation ----------------
__global__ __launch_bounds__(256)
void gates_gemm_kernel(const float* __restrict__ x, const float* __restrict__ hx,
                       const __bf16* __restrict__ w2,
                       const float* __restrict__ bias_ih, const float* __restrict__ bias_hh,
                       float* __restrict__ act_gates) {
    __shared__ __align__(16) __bf16 a_s[GT_TM][LDA];
    __shared__ __align__(16) __bf16 b_s[GT_TN][LDA];
    const int m0   = blockIdx.y * GT_TM;
    const int n0   = blockIdx.x * GT_TN;
    const int tid  = threadIdx.x;
    const int wave = tid >> 5, lane = tid & 31;
    const int half = lane >> 4, l16 = lane & 15;

    f32x8 acc[4] = {};

    for (int kk = 0; kk < K_TOT; kk += GT_TK) {
        // whole 32-wide K slab comes from a single source
        const float* srcA = (kk < IN_DIM) ? (x + kk) : (hx + (kk - IN_DIM));
        stage_a(srcA, m0, a_s, tid);
        stage_b(w2, n0, kk, b_s, tid);
        __syncthreads();

        const __bf16* ap = &a_s[wave * 16 + l16][0];
        bf16x16 afrag = make_frag(ap + 8 * half, ap + 16 + 8 * half);
#pragma unroll
        for (int nt = 0; nt < 4; ++nt) {
            const __bf16* bp = &b_s[nt * 16 + l16][0] + 8 * half;
            bf16x16 bfrag = make_frag(bp, bp + 16);
            acc[nt] = __builtin_amdgcn_wmma_f32_16x16x32_bf16(
                false, afrag, false, bfrag, (short)0, acc[nt], false, false);
        }
        __syncthreads();
    }

#pragma unroll
    for (int nt = 0; nt < 4; ++nt) {
        int n_g = n0 + nt * 16 + l16;
        float bsum = bias_ih[n_g] + bias_hh[n_g];
        float a = ((n_g >> 7) == 2) ? 2.0f : 1.0f;   // tanh for gate 2, sigmoid else
#pragma unroll
        for (int v = 0; v < 8; ++v) {
            int b = m0 + wave * 16 + v + 8 * half;
            act_gates[(size_t)b * GATES + n_g] = gate_act(acc[nt][v] + bsum, a);
        }
    }
}

// ---------------- K3: cy_h = fg*cx + ig*cg  (float4) ----------------
__global__ void cyh_kernel(const float* __restrict__ act_gates,
                           const float* __restrict__ cx,
                           float* __restrict__ cy_h) {
    int i = blockIdx.x * 256 + threadIdx.x;       // one float4 per thread
    if (i >= B_ROWS * H_DIM / 4) return;
    int b = i >> 5, h = (i & 31) * 4;
    const float* ag = act_gates + (size_t)b * GATES;
    f32x4 ig = *(const f32x4*)(ag + h);
    f32x4 fg = *(const f32x4*)(ag + 128 + h);
    f32x4 cg = *(const f32x4*)(ag + 256 + h);
    f32x4 cv = *(const f32x4*)(cx + (size_t)b * H_DIM + h);
    *(f32x4*)(cy_h + (size_t)b * H_DIM + h) = fg * cv + ig * cg;
}

// ---------------- K4: neighborhood gather + softmax attention mix ----------------
__global__ __launch_bounds__(128)
void attention_mix_kernel(const float* __restrict__ memory,
                          const int* __restrict__ grid_idx,
                          const float* __restrict__ cy_h,
                          float* __restrict__ mix) {
    __shared__ float cs_s[4][KNB * H_DIM];   // 51.2 KB
    const int wave = threadIdx.x >> 5, lane = threadIdx.x & 31;
    const int b = blockIdx.x * 4 + wave;
    const int gx = grid_idx[2 * b] + SPW;
    const int gy = grid_idx[2 * b + 1] + SPW;

    float qv[4];
#pragma unroll
    for (int i = 0; i < 4; ++i) qv[i] = cy_h[(size_t)b * H_DIM + lane + 32 * i];

    float s[KNB];
#pragma unroll
    for (int k = 0; k < KNB; ++k) {
        const int dx = k / 5 - SPW, dy = k % 5 - SPW;
        const float* src = memory + ((size_t)(gx + dx) * NMEM + (gy + dy)) * H_DIM;
        if (k + 1 < KNB) {
            const int dx2 = (k + 1) / 5 - SPW, dy2 = (k + 1) % 5 - SPW;
            __builtin_prefetch(memory + ((size_t)(gx + dx2) * NMEM + (gy + dy2)) * H_DIM, 0, 1);
        }
        float p = 0.0f;
#pragma unroll
        for (int i = 0; i < 4; ++i) {
            float v = src[lane + 32 * i];
            cs_s[wave][k * H_DIM + lane + 32 * i] = v;
            p += qv[i] * v;
        }
#pragma unroll
        for (int off = 16; off > 0; off >>= 1) p += __shfl_xor(p, off, 32);
        s[k] = p;
    }
    float mx = s[0];
#pragma unroll
    for (int k = 1; k < KNB; ++k) mx = fmaxf(mx, s[k]);
    float den = 0.0f;
#pragma unroll
    for (int k = 0; k < KNB; ++k) { s[k] = __expf(s[k] - mx); den += s[k]; }
    float inv = __builtin_amdgcn_rcpf(den);

    float mv[4] = {0.f, 0.f, 0.f, 0.f};
#pragma unroll
    for (int k = 0; k < KNB; ++k) {
        float w = s[k] * inv;
#pragma unroll
        for (int i = 0; i < 4; ++i) mv[i] += w * cs_s[wave][k * H_DIM + lane + 32 * i];
    }
#pragma unroll
    for (int i = 0; i < 4; ++i) mix[(size_t)b * H_DIM + lane + 32 * i] = mv[i];
}

// ---------------- K5: attention GEMM (bf16 WMMA) + fused c / hy epilogue ----------------
__global__ __launch_bounds__(256)
void attn_gemm_kernel(const float* __restrict__ mix, const float* __restrict__ cy_h,
                      const __bf16* __restrict__ attn_wbf,
                      const float* __restrict__ attn_b,
                      const float* __restrict__ act_gates,
                      float* __restrict__ out_hy, float* __restrict__ out_c) {
    __shared__ __align__(16) __bf16 a_s[GT_TM][LDA];
    __shared__ __align__(16) __bf16 b_s[GT_TN][LDA];
    const int m0   = blockIdx.y * GT_TM;
    const int n0   = blockIdx.x * GT_TN;     // gridDim.x == 2 -> N=128
    const int tid  = threadIdx.x;
    const int wave = tid >> 5, lane = tid & 31;
    const int half = lane >> 4, l16 = lane & 15;

    f32x8 acc[4] = {};

    for (int kk = 0; kk < K_TOT; kk += GT_TK) {
        const float* srcA = (kk < H_DIM) ? (mix + kk) : (cy_h + (kk - H_DIM));
        stage_a(srcA, m0, a_s, tid);
        stage_b(attn_wbf, n0, kk, b_s, tid);
        __syncthreads();

        const __bf16* ap = &a_s[wave * 16 + l16][0];
        bf16x16 afrag = make_frag(ap + 8 * half, ap + 16 + 8 * half);
#pragma unroll
        for (int nt = 0; nt < 4; ++nt) {
            const __bf16* bp = &b_s[nt * 16 + l16][0] + 8 * half;
            bf16x16 bfrag = make_frag(bp, bp + 16);
            acc[nt] = __builtin_amdgcn_wmma_f32_16x16x32_bf16(
                false, afrag, false, bfrag, (short)0, acc[nt], false, false);
        }
        __syncthreads();
    }

#pragma unroll
    for (int nt = 0; nt < 4; ++nt) {
        int h = n0 + nt * 16 + l16;        // 0..127
        float ab = attn_b[h];
#pragma unroll
        for (int v = 0; v < 8; ++v) {
            int b = m0 + wave * 16 + v + 8 * half;
            float atten = fast_tanh(acc[nt][v] + ab);
            const float* ag = act_gates + (size_t)b * GATES;
            float sg = ag[512 + h], og = ag[384 + h];
            float cy = cy_h[(size_t)b * H_DIM + h];
            float c  = cy + sg * atten;
            out_c[(size_t)b * H_DIM + h]  = c;
            out_hy[(size_t)b * H_DIM + h] = og * fast_tanh(c);
        }
    }
}

// ---------------- K6: streaming copy of memory -> output (non-temporal) ----------------
__global__ void copy_mem_kernel(const float* __restrict__ src, float* __restrict__ dst, long n4) {
    long i = (long)blockIdx.x * blockDim.x + threadIdx.x;
    long stride = (long)gridDim.x * blockDim.x;
    const f32x4* s = (const f32x4*)src;
    f32x4* d = (f32x4*)dst;
    for (; i < n4; i += stride) {
        f32x4 v = __builtin_nontemporal_load(s + i);
        __builtin_nontemporal_store(v, d + i);
    }
}

// ---------------- K7: deterministic scatter (max-b wins for duplicate cells) ----------------
__global__ void fill_owner_kernel(int* __restrict__ owner, int n) {
    int i = blockIdx.x * 256 + threadIdx.x;
    if (i < n) owner[i] = -1;
}
__global__ void claim_kernel(const int* __restrict__ grid_idx, int* __restrict__ owner) {
    int b = blockIdx.x * 256 + threadIdx.x;
    if (b >= B_ROWS) return;
    int gx = grid_idx[2 * b] + SPW, gy = grid_idx[2 * b + 1] + SPW;
    atomicMax(&owner[gx * NMEM + gy], b);
}
__global__ void scatter_write_kernel(const int* __restrict__ grid_idx,
                                     const int* __restrict__ owner,
                                     const float* __restrict__ mem_in,
                                     const float* __restrict__ act_gates,
                                     const float* __restrict__ out_c,
                                     float* __restrict__ out_mem) {
    int i = blockIdx.x * 256 + threadIdx.x;       // one float4 per thread
    if (i >= B_ROWS * H_DIM / 4) return;
    int b = i >> 5, h = (i & 31) * 4;
    int gx = grid_idx[2 * b] + SPW, gy = grid_idx[2 * b + 1] + SPW;
    int cell = gx * NMEM + gy;
    if (owner[cell] != b) return;     // duplicate cells: highest b wins (deterministic)
    f32x4 sg = *(const f32x4*)(act_gates + (size_t)b * GATES + 512 + h);
    size_t off = (size_t)cell * H_DIM + h;
    f32x4 m  = *(const f32x4*)(mem_in + off);
    f32x4 c  = *(const f32x4*)(out_c + (size_t)b * H_DIM + h);
    *(f32x4*)(out_mem + off) = sg * m + (1.0f - sg) * c;
}

// ---------------- launcher ----------------
extern "C" void kernel_launch(void* const* d_in, const int* in_sizes, int n_in,
                              void* d_out, int out_size, void* d_ws, size_t ws_size,
                              hipStream_t stream) {
    const float* x        = (const float*)d_in[0];
    const int*   grid_idx = (const int*)  d_in[1];
    const float* hx       = (const float*)d_in[2];
    const float* cx       = (const float*)d_in[3];
    const float* w_ih     = (const float*)d_in[4];
    const float* w_hh     = (const float*)d_in[5];
    const float* b_ih     = (const float*)d_in[6];
    const float* b_hh     = (const float*)d_in[7];
    const float* attn_w   = (const float*)d_in[8];
    const float* attn_b   = (const float*)d_in[9];
    const float* memory   = (const float*)d_in[10];

    float* out = (float*)d_out;
    float* out_hy  = out;                                   // [B,H]
    float* out_c   = out + (size_t)B_ROWS * H_DIM;          // [B,H]
    float* out_mem = out + 2 * (size_t)B_ROWS * H_DIM;      // [N,N,H]

    // workspace layout (~64 MB)
    char* w = (char*)d_ws;
    __bf16* w2bf      = (__bf16*)w; w += (size_t)GATES * K_TOT * 2;        // 327,680
    __bf16* attn_wbf  = (__bf16*)w; w += (size_t)H_DIM * K_TOT * 2;        //  65,536
    float*  act_gates = (float*)w;  w += (size_t)B_ROWS * GATES * 4;       //  41.9 MB
    float*  cy_h      = (float*)w;  w += (size_t)B_ROWS * H_DIM * 4;       //   8.4 MB
    float*  mixb      = (float*)w;  w += (size_t)B_ROWS * H_DIM * 4;       //   8.4 MB
    int*    owner     = (int*)w;    w += (size_t)NMEM * NMEM * 4;          //   4.9 MB

    convert_weights_kernel<<<(GATES * K_TOT / 4 + 255) / 256, 256, 0, stream>>>(
        w_ih, w_hh, attn_w, w2bf, attn_wbf);

    fill_owner_kernel<<<(NMEM * NMEM + 255) / 256, 256, 0, stream>>>(owner, NMEM * NMEM);

    gates_gemm_kernel<<<dim3(GATES / GT_TN, B_ROWS / GT_TM), 256, 0, stream>>>(
        x, hx, w2bf, b_ih, b_hh, act_gates);

    cyh_kernel<<<(B_ROWS * H_DIM / 4 + 255) / 256, 256, 0, stream>>>(act_gates, cx, cy_h);

    attention_mix_kernel<<<B_ROWS / 4, 128, 0, stream>>>(memory, grid_idx, cy_h, mixb);

    attn_gemm_kernel<<<dim3(H_DIM / GT_TN, B_ROWS / GT_TM), 256, 0, stream>>>(
        mixb, cy_h, attn_wbf, attn_b, act_gates, out_hy, out_c);

    long n4 = (long)NMEM * NMEM * H_DIM / 4;
    copy_mem_kernel<<<8192, 256, 0, stream>>>(memory, out_mem, n4);

    claim_kernel<<<(B_ROWS + 255) / 256, 256, 0, stream>>>(grid_idx, owner);
    scatter_write_kernel<<<(B_ROWS * H_DIM / 4 + 255) / 256, 256, 0, stream>>>(
        grid_idx, owner, memory, act_gates, out_c, out_mem);
}
